// CrossDomainModel_30623116820520
// MI455X (gfx1250) — matrix operand order repaired and checked by reading, not compiled, not run on hardware
//
#include <hip/hip_runtime.h>
#include <math.h>

typedef __attribute__((ext_vector_type(16))) __bf16 v16bf;
typedef __attribute__((ext_vector_type(8)))  __bf16 v8bf;
typedef __attribute__((ext_vector_type(8)))  float  v8f;

// ---------------- problem constants ----------------
constexpr int BATCH = 8;
constexpr int LSAMP = 32000;
constexpr int FFTL  = 256;
constexpr int HOP   = 64;
constexpr int TFRM  = 497;                  // (32000-256)/64+1
constexpr int RROWS = BATCH * TFRM;         // 3976
constexpr int RPAD  = 3984;                 // 249 * 16
constexpr int AE    = 256;
constexpr int NFREQ = 129;
constexpr int FC    = 385;                  // AE + NFREQ
constexpr int FCP   = 416;                  // pad K to mult of 32
constexpr int BTL   = 256;
constexpr int HID   = 512;
constexpr int NBLK  = 32;
constexpr int EMB   = 20;
constexpr int SEPN  = 7700;
constexpr int SEPNP = 7712;                 // 482 * 16
constexpr int R2    = 2 * RROWS;            // 7952 = 497*16
constexpr float FEPS = 1e-12f;
constexpr float PI_F = 3.14159265358979323846f;

__constant__ int d_combs[15][2] = {
  {0,1},{0,2},{0,3},{0,4},{0,5},{1,2},{1,3},{1,4},{1,5},
  {2,3},{2,4},{2,5},{3,4},{3,5},{4,5}};

// ---------------- utility kernels ----------------
__global__ void k_zerof(float* __restrict__ p, size_t n) {
  size_t i = (size_t)blockIdx.x * blockDim.x + threadIdx.x;
  if (i < n) p[i] = 0.0f;
}

__global__ void k_mix(const float* __restrict__ audios, float* __restrict__ mix) {
  size_t i = (size_t)blockIdx.x * blockDim.x + threadIdx.x;
  if (i >= (size_t)BATCH * LSAMP) return;
  size_t b = i / LSAMP, l = i % LSAMP;
  mix[i] = audios[(b * 2 + 0) * (size_t)LSAMP + l] +
           audios[(b * 2 + 1) * (size_t)LSAMP + l];
}

__global__ void k_framesbf(const float* __restrict__ mix, __bf16* __restrict__ fr) {
  size_t i = (size_t)blockIdx.x * blockDim.x + threadIdx.x;
  if (i >= (size_t)RPAD * FFTL) return;
  int k = (int)(i & 255); size_t r = i >> 8;
  __bf16 v = (__bf16)0.0f;
  if (r < (size_t)RROWS) {
    int b = (int)(r / TFRM), t = (int)(r % TFRM);
    v = (__bf16)mix[(size_t)b * LSAMP + t * HOP + k];
  }
  fr[i] = v;
}

// transpose + bf16-pack weight: W[K][N] (fp32) -> Wt[Npad][Kpad] (bf16)
__global__ void k_packw(const float* __restrict__ W, __bf16* __restrict__ Wt,
                        int K, int N, int Kpad, int Npad) {
  size_t i = (size_t)blockIdx.x * blockDim.x + threadIdx.x;
  if (i >= (size_t)Npad * Kpad) return;
  int k = (int)(i % Kpad); int n = (int)(i / Kpad);
  __bf16 v = (__bf16)0.0f;
  if (k < K && n < N) v = (__bf16)W[(size_t)k * N + n];
  Wt[i] = v;
}

// fp32 activations [RPAD x C] -> bf16, zeroing pad rows
__global__ void k_packact(const float* __restrict__ x, __bf16* __restrict__ o, int C) {
  size_t i = (size_t)blockIdx.x * blockDim.x + threadIdx.x;
  if (i >= (size_t)RPAD * C) return;
  size_t r = i / (size_t)C;
  o[i] = (r < (size_t)RROWS) ? (__bf16)x[i] : (__bf16)0.0f;
}

// ---------------- WMMA GEMM (bf16 in, fp32 accum) ----------------
// Each wave computes a 2x4 block of 16x16 tiles (32 rows x 64 cols):
// per K-step of 32: 4 A-loads + 8 B-loads (all B128) feed 8 WMMAs.
// act: 0=none, 1=relu, 2=prelu(alpha per out channel)
__global__ void k_gemm(const __bf16* __restrict__ A,
                       const __bf16* __restrict__ Bt,
                       const float*  __restrict__ bias, int nbias,
                       const float*  __restrict__ alpha,
                       const float*  __restrict__ residual,
                       float*  __restrict__ C,
                       __bf16* __restrict__ Cbf,
                       int Mtiles, int Npad, int Kpad, int act) {
  const int lane = threadIdx.x & 31;
  const int wave = threadIdx.x >> 5;
  const int lmod = lane & 15;
  const int half = lane >> 4;
  const int Ntiles = Npad >> 4;

  const int ntile0 = (blockIdx.x * 4 + wave) * 4;   // 4 N-tiles per wave
  const int mtile0 = blockIdx.y * 2;                // 2 M-tiles per wave

  // A fragment pointers (16-bit 16x32 A layout): lane = row M, VGPR0-3 hold
  // K[8h..8h+7], VGPR4-7 hold K[16+8h..+7] -> two contiguous B128 loads.
  const __bf16* ap[2];
  bool mval[2];
#pragma unroll
  for (int mi = 0; mi < 2; ++mi) {
    int mt = mtile0 + mi;
    mval[mi] = (mt < Mtiles);
    int mc = mval[mi] ? mt : (Mtiles - 1);          // clamp pad tile loads
    ap[mi] = A + (size_t)(mc * 16 + lmod) * Kpad + 8 * half;
  }
  // B fragment pointers (16-bit 32x16 B layout): lane = col N, VGPR0-3 hold
  // K[8h..8h+7], VGPR4-7 hold K[16+8h..+7] -> two contiguous B128 loads.
  const __bf16* bp[4];
  bool nval[4];
#pragma unroll
  for (int ni = 0; ni < 4; ++ni) {
    int nt = ntile0 + ni;
    nval[ni] = (nt < Ntiles);
    int nc = nval[ni] ? nt : 0;                     // clamp pad tile loads
    bp[ni] = Bt + (size_t)(nc * 16 + lmod) * Kpad + 8 * half;
  }

  v8f acc[2][4] = {};
  for (int k = 0; k < Kpad; k += 32) {
    union U { v16bf v; v8bf h[2]; };
    U av[2], bv[4];
#pragma unroll
    for (int mi = 0; mi < 2; ++mi) {
      av[mi].h[0] = *(const v8bf*)(ap[mi] + k);
      av[mi].h[1] = *(const v8bf*)(ap[mi] + k + 16);
    }
#pragma unroll
    for (int ni = 0; ni < 4; ++ni) {
      bv[ni].h[0] = *(const v8bf*)(bp[ni] + k);
      bv[ni].h[1] = *(const v8bf*)(bp[ni] + k + 16);
    }
#pragma unroll
    for (int mi = 0; mi < 2; ++mi)
#pragma unroll
      for (int ni = 0; ni < 4; ++ni)
        acc[mi][ni] = __builtin_amdgcn_wmma_f32_16x16x32_bf16(
            false, av[mi].v, false, bv[ni].v, (short)0, acc[mi][ni],
            false, false);
  }

  // epilogue: bias / activation / residual, masked for edge tiles
#pragma unroll
  for (int ni = 0; ni < 4; ++ni) {
    if (!nval[ni]) continue;
    const int gcol = (ntile0 + ni) * 16 + lmod;
    const float bb = (bias != nullptr && gcol < nbias) ? bias[gcol] : 0.0f;
    const float al = (act == 2) ? alpha[gcol] : 0.0f;
#pragma unroll
    for (int mi = 0; mi < 2; ++mi) {
      if (!mval[mi]) continue;
      const int rbase = (mtile0 + mi) * 16 + half * 8;
#pragma unroll
      for (int v = 0; v < 8; ++v) {
        float x = acc[mi][ni][v] + bb;
        if (act == 1)      x = fmaxf(x, 0.0f);
        else if (act == 2) x = fmaxf(x, 0.0f) + al * fminf(x, 0.0f);
        size_t idx = (size_t)(rbase + v) * Npad + gcol;
        if (residual) x += residual[idx];
        if (Cbf) Cbf[idx] = (__bf16)x;
        else     C[idx]   = x;
      }
    }
  }
}

// ---------------- STFT (direct 256-pt rDFT) ----------------
__global__ void k_fft(const float* __restrict__ mix,
                      float* __restrict__ mag, float* __restrict__ phase) {
  size_t i = (size_t)blockIdx.x * blockDim.x + threadIdx.x;
  if (i >= (size_t)RROWS * NFREQ) return;
  int kk = (int)(i % NFREQ); size_t r = i / NFREQ;
  int b = (int)(r / TFRM), t = (int)(r % TFRM);
  const float* xp = mix + (size_t)b * LSAMP + t * HOP;
  const float w = -2.0f * PI_F * (float)kk / (float)FFTL;
  float re = 0.0f, im = 0.0f;
  for (int j = 0; j < FFTL; ++j) {
    float win = sqrtf(0.5f - 0.5f * cosf(2.0f * PI_F * (float)j / (float)FFTL));
    float x = xp[j] * win;
    float s, c; sincosf(w * (float)j, &s, &c);
    re += x * c; im += x * s;
  }
  mag[i]   = sqrtf(re * re + im * im);
  phase[i] = atan2f(im, re);
}

// ---------------- channel layernorm over concat[enc, log1p(mag)] ----------------
__global__ void k_cln(const float* __restrict__ enc, const float* __restrict__ mag,
                      const float* __restrict__ g, const float* __restrict__ bta,
                      __bf16* __restrict__ outbf) {
  int r = blockIdx.x;            // [0, RPAD)
  int tid = threadIdx.x;         // 128
  __shared__ float ssum[128], ssq[128];
  float vals[4];
  float s = 0.0f, q = 0.0f;
#pragma unroll
  for (int i = 0; i < 4; ++i) {
    int j = tid + i * 128;
    float v = 0.0f;
    if (r < RROWS && j < FC)
      v = (j < AE) ? enc[(size_t)r * AE + j]
                   : log1pf(mag[(size_t)r * NFREQ + (j - AE)]);
    vals[i] = v;
    if (j < FC) { s += v; q += v * v; }
  }
  ssum[tid] = s; ssq[tid] = q;
  __syncthreads();
  for (int off = 64; off > 0; off >>= 1) {
    if (tid < off) { ssum[tid] += ssum[tid + off]; ssq[tid] += ssq[tid + off]; }
    __syncthreads();
  }
  float m   = ssum[0] / (float)FC;
  float var = ssq[0] / (float)FC - m * m;
  float inv = rsqrtf(var + FEPS);
#pragma unroll
  for (int i = 0; i < 4; ++i) {
    int j = tid + i * 128;
    if (j < FCP) {
      float o = 0.0f;
      if (r < RROWS && j < FC) o = (vals[i] - m) * inv * g[j] + bta[j];
      outbf[(size_t)r * FCP + j] = (__bf16)o;
    }
  }
}

// ---------------- gLN statistics (per batch over T x HID) ----------------
__global__ void k_stats(const float* __restrict__ y, float* __restrict__ stats) {
  const int b = blockIdx.y;
  const size_t n = (size_t)TFRM * HID;
  const size_t base = (size_t)b * n;
  float s = 0.0f, q = 0.0f;
  for (size_t i = (size_t)blockIdx.x * blockDim.x + threadIdx.x; i < n;
       i += (size_t)gridDim.x * blockDim.x) {
    float v = y[base + i]; s += v; q += v * v;
  }
  __shared__ float ss[256], sq[256];
  int tid = threadIdx.x;
  ss[tid] = s; sq[tid] = q;
  __syncthreads();
  for (int off = 128; off > 0; off >>= 1) {
    if (tid < off) { ss[tid] += ss[tid + off]; sq[tid] += sq[tid + off]; }
    __syncthreads();
  }
  if (tid == 0) {
    atomicAdd(&stats[b * 2 + 0], ss[0]);
    atomicAdd(&stats[b * 2 + 1], sq[0]);
  }
}

// ---------------- gLN-apply + dilated depthwise conv + PReLU2 ----------------
__global__ void k_dwconv(const float* __restrict__ y, const float* __restrict__ stats,
                         const float* __restrict__ g1, const float* __restrict__ b1,
                         const float* __restrict__ dwW, const float* __restrict__ p2,
                         float* __restrict__ z, int di) {
  size_t i = (size_t)blockIdx.x * blockDim.x + threadIdx.x;
  if (i >= (size_t)RROWS * HID) return;
  int c = (int)(i & (HID - 1));
  size_t r = i >> 9;
  int b = (int)(r / TFRM), t = (int)(r % TFRM);
  const float invN = 1.0f / ((float)TFRM * (float)HID);
  float m   = stats[b * 2 + 0] * invN;
  float var = stats[b * 2 + 1] * invN - m * m;
  float inv = rsqrtf(var + FEPS);
  float acc = 0.0f;
#pragma unroll
  for (int k = 0; k < 3; ++k) {
    int tt = t + (k - 1) * di;
    if (tt >= 0 && tt < TFRM) {
      float v = y[((size_t)b * TFRM + tt) * HID + c];
      v = (v - m) * inv * g1[c] + b1[c];
      acc += dwW[k * HID + c] * v;
    }
  }
  float a = p2[c];
  z[i] = fmaxf(acc, 0.0f) + a * fminf(acc, 0.0f);
}

// ---------------- gLN-apply + bf16 pack (for c2 GEMM) ----------------
__global__ void k_glnpack(const float* __restrict__ z, const float* __restrict__ stats,
                          const float* __restrict__ g, const float* __restrict__ bb,
                          __bf16* __restrict__ o) {
  size_t i = (size_t)blockIdx.x * blockDim.x + threadIdx.x;
  if (i >= (size_t)RPAD * HID) return;
  int c = (int)(i & (HID - 1));
  size_t r = i >> 9;
  float out = 0.0f;
  if (r < (size_t)RROWS) {
    int b = (int)(r / TFRM);
    const float invN = 1.0f / ((float)TFRM * (float)HID);
    float m   = stats[b * 2 + 0] * invN;
    float var = stats[b * 2 + 1] * invN - m * m;
    float inv = rsqrtf(var + FEPS);
    out = (z[i] - m) * inv * g[c] + bb[c];
  }
  o[i] = (__bf16)out;
}

// ---------------- attractor assignment accumulation ----------------
__global__ void k_assign(const __bf16* __restrict__ emb, const float* __restrict__ anchors,
                         float* __restrict__ num, float* __restrict__ den) {
  int bp = blockIdx.y;                 // B * 15
  int b = bp / 15, p = bp % 15;
  int i0 = d_combs[p][0], i1 = d_combs[p][1];
  float a0[EMB], a1[EMB];
#pragma unroll
  for (int e = 0; e < EMB; ++e) { a0[e] = anchors[i0 * EMB + e]; a1[e] = anchors[i1 * EMB + e]; }
  float n0[EMB] = {}, n1[EMB] = {};
  float dd0 = 0.0f, dd1 = 0.0f;
  const int total = TFRM * FC;
  for (int i = blockIdx.x * blockDim.x + threadIdx.x; i < total;
       i += gridDim.x * blockDim.x) {
    int t = i / FC, f = i % FC;
    const __bf16* ep = emb + ((size_t)b * TFRM + t) * SEPNP + f * EMB;
    float ev[EMB]; float s0 = 0.0f, s1 = 0.0f;
#pragma unroll
    for (int e = 0; e < EMB; ++e) {
      float v = (float)ep[e]; ev[e] = v;
      s0 += v * a0[e]; s1 += v * a1[e];
    }
    float mx = fmaxf(s0, s1);
    float e0 = expf(s0 - mx), e1 = expf(s1 - mx);
    float inv = 1.0f / (e0 + e1);
    float p0 = e0 * inv, p1 = e1 * inv;
    dd0 += p0; dd1 += p1;
#pragma unroll
    for (int e = 0; e < EMB; ++e) { n0[e] += p0 * ev[e]; n1[e] += p1 * ev[e]; }
  }
  __shared__ float sacc[42];
  if (threadIdx.x < 42) sacc[threadIdx.x] = 0.0f;
  __syncthreads();
#pragma unroll
  for (int e = 0; e < EMB; ++e) {
    atomicAdd(&sacc[e], n0[e]);
    atomicAdd(&sacc[EMB + e], n1[e]);
  }
  atomicAdd(&sacc[40], dd0); atomicAdd(&sacc[41], dd1);
  __syncthreads();
  if (threadIdx.x < 42) {
    float v = sacc[threadIdx.x];
    if (threadIdx.x < EMB)
      atomicAdd(&num[(((size_t)b * 15 + p) * 2 + 0) * EMB + threadIdx.x], v);
    else if (threadIdx.x < 2 * EMB)
      atomicAdd(&num[(((size_t)b * 15 + p) * 2 + 1) * EMB + (threadIdx.x - EMB)], v);
    else
      atomicAdd(&den[((size_t)b * 15 + p) * 2 + (threadIdx.x - 40)], v);
  }
}

// ---------------- attractor selection (tiny) ----------------
__global__ void k_attract(const float* __restrict__ num, const float* __restrict__ den,
                          float* __restrict__ attractors) {
  int b = threadIdx.x;
  if (b >= BATCH) return;
  float best = 3.0e38f;
  float at[2][EMB];
  for (int p = 0; p < 15; ++p) {
    float a[2][EMB];
    for (int c = 0; c < 2; ++c) {
      float d = den[((size_t)b * 15 + p) * 2 + c];
      float invd = 1.0f / d;
      for (int e = 0; e < EMB; ++e)
        a[c][e] = num[(((size_t)b * 15 + p) * 2 + c) * EMB + e] * invd;
    }
    float s = 0.0f;
    for (int e = 0; e < EMB; ++e) s += a[0][e] * a[1][e];
    if (s < best) {
      best = s;
      for (int c = 0; c < 2; ++c)
        for (int e = 0; e < EMB; ++e) at[c][e] = a[c][e];
    }
  }
  for (int c = 0; c < 2; ++c)
    for (int e = 0; e < EMB; ++e)
      attractors[((size_t)b * 2 + c) * EMB + e] = at[c][e];
}

// ---------------- code = (emb . attractor) * feat ----------------
__global__ void k_code(const __bf16* __restrict__ emb, const float* __restrict__ attractors,
                       const float* __restrict__ enc, const float* __restrict__ mag,
                       __bf16* __restrict__ codebf, float* __restrict__ sm) {
  size_t i = (size_t)blockIdx.x * blockDim.x + threadIdx.x;
  if (i >= (size_t)BATCH * 2 * TFRM * FC) return;
  int f = (int)(i % FC); size_t rem = i / FC;
  int t = (int)(rem % TFRM); rem /= TFRM;
  int c = (int)(rem % 2);  int b = (int)(rem / 2);
  size_t r = (size_t)b * TFRM + t;
  const __bf16* ep = emb + r * SEPNP + f * EMB;
  const float* at = attractors + ((size_t)b * 2 + c) * EMB;
  float s = 0.0f;
#pragma unroll
  for (int e = 0; e < EMB; ++e) s += (float)ep[e] * at[e];
  float feat = (f < AE) ? enc[r * AE + f] : mag[r * NFREQ + (f - AE)];
  float val = s * feat;
  size_t r2 = ((size_t)b * 2 + c) * TFRM + t;
  if (f < AE) codebf[r2 * AE + f] = (__bf16)val;
  else        sm[r2 * NFREQ + (f - AE)] = val;
}

// ---------------- iSTFT frames (direct iDFT) * INV_WIN ----------------
__global__ void k_istft(const float* __restrict__ sm, const float* __restrict__ phase,
                        float* __restrict__ istfr) {
  size_t i = (size_t)blockIdx.x * blockDim.x + threadIdx.x;
  if (i >= (size_t)R2 * FFTL) return;
  int n = (int)(i & 255); size_t r2 = i >> 8;
  int t = (int)(r2 % TFRM); size_t bc = r2 / TFRM; int b = (int)(bc / 2);
  size_t rp = ((size_t)b * TFRM + t) * NFREQ;
  const float* smr = sm + r2 * NFREQ;
  float acc = cosf(phase[rp + 0]) * smr[0];        // k = 0 (real part)
  const float w0 = 2.0f * PI_F * (float)n / (float)FFTL;
  for (int k = 1; k < 128; ++k) {
    float s, c; sincosf(phase[rp + k], &s, &c);
    float re = c * smr[k], im = s * smr[k];
    float st, ct; sincosf(w0 * (float)k, &st, &ct);
    acc += 2.0f * (re * ct - im * st);
  }
  acc += cosf(phase[rp + 128]) * smr[128] * ((n & 1) ? -1.0f : 1.0f);  // k = 128
  acc *= (1.0f / (float)FFTL);
  // INV_WIN[n] = WIN[n] / sum_r WIN[(n%64)+64r]^2
  float win = sqrtf(0.5f - 0.5f * cosf(2.0f * PI_F * (float)n / (float)FFTL));
  int nm = n & 63; float denom = 0.0f;
#pragma unroll
  for (int rr = 0; rr < 4; ++rr)
    denom += 0.5f - 0.5f * cosf(2.0f * PI_F * (float)(nm + 64 * rr) / (float)FFTL);
  istfr[i] = acc * (win / denom);
}

// ---------------- overlap-add + 50/50 mix ----------------
__global__ void k_ola(const float* __restrict__ decfr, const float* __restrict__ istfr,
                      float* __restrict__ out) {
  size_t i = (size_t)blockIdx.x * blockDim.x + threadIdx.x;
  if (i >= (size_t)BATCH * 2 * LSAMP) return;
  int l = (int)(i % LSAMP); size_t bc = i / LSAMP;
  int t0 = (l >= FFTL) ? ((l - FFTL + 1 + HOP - 1) / HOP) : 0;
  int t1 = l / HOP; if (t1 >= TFRM) t1 = TFRM - 1;
  float ac = 0.0f, ai = 0.0f;
  for (int t = t0; t <= t1; ++t) {
    int j = l - t * HOP;
    size_t idx = (bc * TFRM + t) * (size_t)FFTL + j;
    ac += decfr[idx]; ai += istfr[idx];
  }
  out[i] = 0.5f * ac + 0.5f * ai;
}

// ---------------- host orchestration ----------------
extern "C" void kernel_launch(void* const* d_in, const int* in_sizes, int n_in,
                              void* d_out, int out_size, void* d_ws, size_t ws_size,
                              hipStream_t stream) {
  (void)in_sizes; (void)n_in; (void)out_size; (void)ws_size;
  const float* audios   = (const float*)d_in[0];
  const float* enc_w    = (const float*)d_in[1];
  const float* enc_b    = (const float*)d_in[2];
  const float* bot_g    = (const float*)d_in[3];
  const float* bot_beta = (const float*)d_in[4];
  const float* bot_w    = (const float*)d_in[5];
  const float* bot_b    = (const float*)d_in[6];
  const float* c1_w     = (const float*)d_in[7];
  const float* c1_b     = (const float*)d_in[8];
  const float* p1       = (const float*)d_in[9];
  const float* g1g      = (const float*)d_in[10];
  const float* g1b      = (const float*)d_in[11];
  const float* dwW      = (const float*)d_in[12];
  const float* p2       = (const float*)d_in[13];
  const float* g2g      = (const float*)d_in[14];
  const float* g2b      = (const float*)d_in[15];
  const float* c2_w     = (const float*)d_in[16];
  const float* c2_b     = (const float*)d_in[17];
  const float* sep_w    = (const float*)d_in[18];
  const float* sep_b    = (const float*)d_in[19];
  const float* anchors  = (const float*)d_in[20];
  const float* dec_w    = (const float*)d_in[21];
  const float* dec_b    = (const float*)d_in[22];
  float* out = (float*)d_out;

  char* base = (char*)d_ws;
  size_t off = 0;
  auto alloc = [&](size_t bytes) -> char* {
    char* p = base + off;
    off = (off + bytes + 255) & ~(size_t)255;
    return p;
  };
  float*  mix       = (float*)alloc(sizeof(float) * BATCH * LSAMP);
  __bf16* framesbf  = (__bf16*)alloc(sizeof(__bf16) * (size_t)RPAD * FFTL);
  float*  enc       = (float*)alloc(sizeof(float) * (size_t)RPAD * AE);
  float*  mag       = (float*)alloc(sizeof(float) * (size_t)RROWS * NFREQ);
  float*  phase     = (float*)alloc(sizeof(float) * (size_t)RROWS * NFREQ);
  __bf16* actbf     = (__bf16*)alloc(sizeof(__bf16) * (size_t)RPAD * HID);
  float*  xbuf      = (float*)alloc(sizeof(float) * (size_t)RPAD * BTL);
  float*  ybuf      = (float*)alloc(sizeof(float) * (size_t)RPAD * HID);
  float*  zbuf      = (float*)alloc(sizeof(float) * (size_t)RPAD * HID);
  float*  stats     = (float*)alloc(sizeof(float) * 16);
  float*  attrnum   = (float*)alloc(sizeof(float) * BATCH * 15 * 2 * EMB);
  float*  attrden   = (float*)alloc(sizeof(float) * BATCH * 15 * 2);
  float*  attractrs = (float*)alloc(sizeof(float) * BATCH * 2 * EMB);
  __bf16* enc_wt    = (__bf16*)alloc(sizeof(__bf16) * 256 * 256);
  __bf16* bottle_wt = (__bf16*)alloc(sizeof(__bf16) * 256 * FCP);
  __bf16* c1_wt     = (__bf16*)alloc(sizeof(__bf16) * (size_t)NBLK * HID * BTL);
  __bf16* c2_wt     = (__bf16*)alloc(sizeof(__bf16) * (size_t)NBLK * BTL * HID);
  __bf16* sep_wt    = (__bf16*)alloc(sizeof(__bf16) * (size_t)SEPNP * BTL);
  __bf16* dec_wt    = (__bf16*)alloc(sizeof(__bf16) * 256 * 256);
  __bf16* embbf     = (__bf16*)alloc(sizeof(__bf16) * (size_t)RPAD * SEPNP);
  __bf16* codebf    = (__bf16*)alloc(sizeof(__bf16) * (size_t)R2 * AE);
  float*  smbuf     = (float*)alloc(sizeof(float) * (size_t)R2 * NFREQ);
  float*  decfr     = (float*)alloc(sizeof(float) * (size_t)R2 * FFTL);
  float*  istfr     = (float*)alloc(sizeof(float) * (size_t)R2 * FFTL);

  auto g1 = [](size_t n, int b) { return dim3((unsigned)((n + b - 1) / b)); };
  // grid: each block = 4 waves x (2M x 4N tiles) => 32 rows x 256 cols / block
  auto gemm = [&](const __bf16* A, const __bf16* Bt, const float* bias, int nbias,
                  const float* alpha, const float* res, float* C, __bf16* Cbf,
                  int Mtiles, int Npad, int Kpad, int act) {
    int ntiles = Npad / 16;
    dim3 grid((unsigned)((ntiles + 15) / 16), (unsigned)((Mtiles + 1) / 2));
    k_gemm<<<grid, 128, 0, stream>>>(A, Bt, bias, nbias, alpha, res, C, Cbf,
                                     Mtiles, Npad, Kpad, act);
  };
  auto packw = [&](const float* W, __bf16* Wt, int K, int N, int Kpad, int Npad) {
    size_t n = (size_t)Npad * Kpad;
    k_packw<<<g1(n, 256), 256, 0, stream>>>(W, Wt, K, N, Kpad, Npad);
  };

  // ---- weight packing (bf16, transposed, padded) ----
  packw(enc_w, enc_wt, 256, 256, 256, 256);
  packw(bot_w, bottle_wt, FC, BTL, FCP, BTL);
  for (int i = 0; i < NBLK; ++i) {
    packw(c1_w + (size_t)i * BTL * HID, c1_wt + (size_t)i * HID * BTL, BTL, HID, BTL, HID);
    packw(c2_w + (size_t)i * HID * BTL, c2_wt + (size_t)i * BTL * HID, HID, BTL, HID, BTL);
  }
  packw(sep_w, sep_wt, BTL, SEPN, BTL, SEPNP);
  packw(dec_w, dec_wt, 256, 256, 256, 256);

  // ---- front end ----
  k_mix<<<g1((size_t)BATCH * LSAMP, 256), 256, 0, stream>>>(audios, mix);
  k_framesbf<<<g1((size_t)RPAD * FFTL, 256), 256, 0, stream>>>(mix, framesbf);
  gemm(framesbf, enc_wt, enc_b, 256, nullptr, nullptr, enc, nullptr,
       RPAD / 16, 256, 256, /*relu*/1);
  k_fft<<<g1((size_t)RROWS * NFREQ, 128), 128, 0, stream>>>(mix, mag, phase);
  k_cln<<<dim3(RPAD), 128, 0, stream>>>(enc, mag, bot_g, bot_beta, actbf);
  gemm(actbf, bottle_wt, bot_b, BTL, nullptr, nullptr, xbuf, nullptr,
       RPAD / 16, BTL, FCP, 0);

  // ---- 32 TCN blocks ----
  for (int i = 0; i < NBLK; ++i) {
    int di = 1 << (i % 8);
    k_packact<<<g1((size_t)RPAD * BTL, 256), 256, 0, stream>>>(xbuf, actbf, BTL);
    gemm(actbf, c1_wt + (size_t)i * HID * BTL, c1_b + (size_t)i * HID, HID,
         p1 + (size_t)i * HID, nullptr, ybuf, nullptr, RPAD / 16, HID, BTL, /*prelu*/2);
    k_zerof<<<1, 32, 0, stream>>>(stats, 16);
    k_stats<<<dim3(48, BATCH), 256, 0, stream>>>(ybuf, stats);
    k_dwconv<<<g1((size_t)RROWS * HID, 256), 256, 0, stream>>>(
        ybuf, stats, g1g + (size_t)i * HID, g1b + (size_t)i * HID,
        dwW + (size_t)i * 3 * HID, p2 + (size_t)i * HID, zbuf, di);
    k_zerof<<<1, 32, 0, stream>>>(stats, 16);
    k_stats<<<dim3(48, BATCH), 256, 0, stream>>>(zbuf, stats);
    k_glnpack<<<g1((size_t)RPAD * HID, 256), 256, 0, stream>>>(
        zbuf, stats, g2g + (size_t)i * HID, g2b + (size_t)i * HID, actbf);
    gemm(actbf, c2_wt + (size_t)i * BTL * HID, c2_b + (size_t)i * BTL, BTL,
         nullptr, xbuf, xbuf, nullptr, RPAD / 16, BTL, HID, 0);
  }

  // ---- separation embedding ----
  k_packact<<<g1((size_t)RPAD * BTL, 256), 256, 0, stream>>>(xbuf, actbf, BTL);
  gemm(actbf, sep_wt, sep_b, SEPN, nullptr, nullptr, nullptr, embbf,
       RPAD / 16, SEPNP, BTL, 0);

  // ---- attractors ----
  k_zerof<<<g1(BATCH * 15 * 2 * (EMB + 1), 256), 256, 0, stream>>>(
      attrnum, (size_t)BATCH * 15 * 2 * EMB);
  k_zerof<<<1, 256, 0, stream>>>(attrden, (size_t)BATCH * 15 * 2);
  k_assign<<<dim3(64, BATCH * 15), 256, 0, stream>>>(embbf, anchors, attrnum, attrden);
  k_attract<<<1, 64, 0, stream>>>(attrnum, attrden, attractrs);

  // ---- masked code, decoder, iSTFT, OLA ----
  k_code<<<g1((size_t)BATCH * 2 * TFRM * FC, 256), 256, 0, stream>>>(
      embbf, attractrs, enc, mag, codebf, smbuf);
  gemm(codebf, dec_wt, dec_b, 256, nullptr, nullptr, decfr, nullptr,
       R2 / 16, 256, 256, 0);
  k_istft<<<g1((size_t)R2 * FFTL, 128), 128, 0, stream>>>(smbuf, phase, istfr);
  k_ola<<<g1((size_t)BATCH * 2 * LSAMP, 256), 256, 0, stream>>>(decfr, istfr, out);
}
